// Classifier_28467043238681
// MI455X (gfx1250) — matrix operand order repaired and checked by reading, not compiled, not run on hardware
//
#include <hip/hip_runtime.h>
#include <hip/hip_bf16.h>
#include <math.h>

// ---------------------------------------------------------------------------
// Problem constants (from reference): B=4096, N=10000, D=512
// ---------------------------------------------------------------------------
#define B_DIM 4096
#define N_DIM 10000
#define D_DIM 512
#define NTILES 625            // N / 16

typedef _Float16 v8h  __attribute__((ext_vector_type(8)));
typedef _Float16 v16h __attribute__((ext_vector_type(16)));
typedef float    v8f  __attribute__((ext_vector_type(8)));

// ---------------------------------------------------------------------------
// Kernel 1: row-wise L2 normalize (f32 in) -> f16 out.  One block per row,
// 256 threads, 2 elements each (D=512). wave32 shfl reduction + LDS combine.
// ---------------------------------------------------------------------------
__global__ void normalize_rows_f16(const float* __restrict__ in,
                                   _Float16* __restrict__ out) {
    __shared__ float red[8];
    const int row = blockIdx.x;
    const int tid = threadIdx.x;
    const float* r = in + (size_t)row * D_DIM;

    float v0 = r[tid];
    float v1 = r[tid + 256];
    float s = v0 * v0 + v1 * v1;

    // wave32 reduction
    #pragma unroll
    for (int off = 16; off > 0; off >>= 1)
        s += __shfl_xor(s, off, 32);
    if ((tid & 31) == 0) red[tid >> 5] = s;
    __syncthreads();
    if (tid == 0) {
        float t = 0.0f;
        #pragma unroll
        for (int i = 0; i < 8; ++i) t += red[i];
        red[0] = __frsqrt_rn(t + 1e-12f);
    }
    __syncthreads();
    const float scale = red[0];

    _Float16* o = out + (size_t)row * D_DIM;
    o[tid]       = (_Float16)(v0 * scale);
    o[tid + 256] = (_Float16)(v1 * scale);
}

// ---------------------------------------------------------------------------
// Kernel 2: WMMA GEMM  cos = Xn[B,D] * En[N,D]^T,  then logits = 1 - dist.
//
// Block: 256 threads = 8 wave32s. blockIdx.y selects a 16-row M tile,
// blockIdx.x * 32 + wave*4 selects four 16-col N tiles per wave.
// Each wave: 4 f32 accumulators (16x16), K loop of 16 steps of
// v_wmma_f32_16x16x32_f16, A fragment reused across the 4 B tiles.
//
// Fragment loads follow the CDNA5 16-bit layouts (ISA 7.12.2):
//   A (16x32): lane L holds row (L&15); lanes<16 get K {k..k+7, k+16..k+23},
//              lanes>=16 get K {k+8..k+15, k+24..k+31}  -> two v8h loads.
//   B (32x16): lane L holds col (L&15); same K split as A:
//              lanes<16 {k..k+7, k+16..k+23}, lanes>=16 {k+8..k+15, k+24..k+31} -> two v8h loads.
// Both operands come straight from row-major f16 (row = D contiguous halfs).
// ---------------------------------------------------------------------------
__global__ void __launch_bounds__(256)
gemm_spherical_logits(const _Float16* __restrict__ xn,
                      const _Float16* __restrict__ en,
                      float* __restrict__ out) {
    const int tid  = threadIdx.x;
    const int wave = tid >> 5;
    const int lane = tid & 31;
    const int lane_lo = lane & 15;
    const int lane_hi = lane >> 4;

    const int m0    = blockIdx.y << 4;               // 256 M tiles
    const int tbase = blockIdx.x * 32 + wave * 4;    // first of 4 N tiles

    // A: pointer pre-offset by the lane's K sub-chunk (8 halfs for lane_hi=1)
    const _Float16* arow =
        xn + (size_t)(m0 + lane_lo) * D_DIM + (lane_hi << 3);

    // B: 4 tiles, clamped to last valid tile so WMMA always runs full-EXEC
    bool tvalid[4];
    int  tcol[4];
    const _Float16* brow[4];
    #pragma unroll
    for (int j = 0; j < 4; ++j) {
        int t = tbase + j;
        tvalid[j] = (t < NTILES);
        if (t >= NTILES) t = NTILES - 1;
        tcol[j] = t << 4;
        brow[j] = en + (size_t)(tcol[j] + lane_lo) * D_DIM + (lane_hi << 3);
    }

    v8f acc[4] = {};

    #pragma unroll 4
    for (int k = 0; k < D_DIM; k += 32) {
        v8h alo = *(const v8h*)(arow + k);
        v8h ahi = *(const v8h*)(arow + k + 16);
        v16h af = __builtin_shufflevector(alo, ahi,
            0, 1, 2, 3, 4, 5, 6, 7, 8, 9, 10, 11, 12, 13, 14, 15);

        #pragma unroll
        for (int j = 0; j < 4; ++j) {
            v8h blo = *(const v8h*)(brow[j] + k);
            v8h bhi = *(const v8h*)(brow[j] + k + 16);
            v16h bf = __builtin_shufflevector(blo, bhi,
                0, 1, 2, 3, 4, 5, 6, 7, 8, 9, 10, 11, 12, 13, 14, 15);

            acc[j] = __builtin_amdgcn_wmma_f32_16x16x32_f16(
                /*neg_a=*/false, af, /*neg_b=*/false, bf,
                /*c_mod=*/(short)0, acc[j],
                /*reuse_a=*/false, /*reuse_b=*/false);
        }
    }

    // Epilogue: spherical distance -> logit, scatter per C-layout
    // (VGPR r: lanes 0-15 -> M=m0+r, lanes 16-31 -> M=m0+r+8; N = tcol+lane_lo)
    const int rowbase = m0 + (lane_hi << 3);
    #pragma unroll
    for (int j = 0; j < 4; ++j) {
        if (!tvalid[j]) continue;                 // wave-uniform branch
        const int col = tcol[j] + lane_lo;
        #pragma unroll
        for (int r = 0; r < 8; ++r) {
            float cosv = acc[j][r];
            float r2   = fminf(fmaxf((1.0f - cosv) * 0.5f, 0.0f), 1.0f);
            float rad  = sqrtf(r2);
            float as   = asinf(rad);
            float logit = 1.0f - 2.0f * as * as;
            out[(size_t)(rowbase + r) * N_DIM + col] = logit;
        }
    }
}

// ---------------------------------------------------------------------------
// Kernel 3: in-place row softmax over N=10000. One block (256 thr) per row,
// values cached in registers (<= 40 per thread), shfl + LDS reductions.
// ---------------------------------------------------------------------------
__global__ void __launch_bounds__(256)
softmax_rows(float* __restrict__ data) {
    __shared__ float red[8];
    __shared__ float bcast;
    const int row = blockIdx.x;
    const int tid = threadIdx.x;
    float* p = data + (size_t)row * N_DIM;

    float vals[40];
    int cnt = 0;
    float mx = -1e30f;
    for (int j = tid; j < N_DIM; j += 256) {
        float v = p[j];
        vals[cnt++] = v;
        mx = fmaxf(mx, v);
    }

    // block max
    #pragma unroll
    for (int off = 16; off > 0; off >>= 1)
        mx = fmaxf(mx, __shfl_xor(mx, off, 32));
    if ((tid & 31) == 0) red[tid >> 5] = mx;
    __syncthreads();
    if (tid == 0) {
        float t = red[0];
        #pragma unroll
        for (int i = 1; i < 8; ++i) t = fmaxf(t, red[i]);
        bcast = t;
    }
    __syncthreads();
    mx = bcast;

    // exp + block sum
    float sum = 0.0f;
    for (int i = 0; i < cnt; ++i) {
        vals[i] = __expf(vals[i] - mx);
        sum += vals[i];
    }
    #pragma unroll
    for (int off = 16; off > 0; off >>= 1)
        sum += __shfl_xor(sum, off, 32);
    if ((tid & 31) == 0) red[tid >> 5] = sum;
    __syncthreads();
    if (tid == 0) {
        float t = 0.0f;
        #pragma unroll
        for (int i = 0; i < 8; ++i) t += red[i];
        bcast = t;
    }
    __syncthreads();
    const float inv = 1.0f / bcast;

    cnt = 0;
    for (int j = tid; j < N_DIM; j += 256)
        p[j] = vals[cnt++] * inv;
}

// ---------------------------------------------------------------------------
// Launch
// ---------------------------------------------------------------------------
extern "C" void kernel_launch(void* const* d_in, const int* in_sizes, int n_in,
                              void* d_out, int out_size, void* d_ws, size_t ws_size,
                              hipStream_t stream) {
    (void)in_sizes; (void)n_in; (void)out_size; (void)ws_size;

    const float* x   = (const float*)d_in[0];   // [4096, 512] f32
    const float* emb = (const float*)d_in[1];   // [10000, 512] f32
    float* out = (float*)d_out;                 // [4096, 10000] f32

    _Float16* xn = (_Float16*)d_ws;                                  // 4 MB
    _Float16* en = (_Float16*)((char*)d_ws + (size_t)B_DIM * D_DIM * sizeof(_Float16));

    normalize_rows_f16<<<B_DIM, 256, 0, stream>>>(x,   xn);
    normalize_rows_f16<<<N_DIM, 256, 0, stream>>>(emb, en);

    // 625 N-tiles / (8 waves * 4 tiles) = 19.5 -> 20 blocks in x
    dim3 grid((NTILES + 31) / 32, B_DIM / 16);
    gemm_spherical_logits<<<grid, 256, 0, stream>>>(xn, en, out);

    softmax_rows<<<B_DIM, 256, 0, stream>>>(out);
}
